// graphNetworkEmbbed_12034498363552
// MI455X (gfx1250) — matrix operand order repaired and checked by reading, not compiled, not run on hardware
//
#include <hip/hip_runtime.h>
#include <hip/hip_bf16.h>
#include <math.h>

typedef _Float16 v16h __attribute__((ext_vector_type(16)));
typedef _Float16 v8h  __attribute__((ext_vector_type(8)));
typedef float    v8f  __attribute__((ext_vector_type(8)));

#define LDT 40   // f16 LDS row stride (80B): 16B-aligned chunks, skewed banks

// ---------------- CDNA5 async global->LDS (raw 16B per lane, ASYNCcnt) ----------------
// ISA 10.2: generic shared-aperture address low 32 bits == LDS byte address, so the
// truncated generic pointer is a valid LDS-address VGPR for the async op.
static __device__ __forceinline__ void async_load_b128(void* lds_dst, const void* gsrc) {
    unsigned ldsaddr = (unsigned)(unsigned long long)lds_dst;
    asm volatile("global_load_async_to_lds_b128 %0, %1, off"
                 :: "v"(ldsaddr), "v"(gsrc) : "memory");
}
static __device__ __forceinline__ void wait_async0() {
    asm volatile("s_wait_asynccnt 0" ::: "memory");
}

static __device__ __forceinline__ v16h cat8(v8h lo, v8h hi) {
    return __builtin_shufflevector(lo, hi, 0,1,2,3,4,5,6,7,8,9,10,11,12,13,14,15);
}

// A-fragment (16x32 f16) built from an f32 LDS tile [m][32] (compact, 128B rows).
// lo lanes: K0..7 / K16..23  hi lanes: K8..15 / K24..31  (ISA 7.12.2 layout)
static __device__ __forceinline__ v16h afrag_from_f32(const float* tile, int row0, int lane) {
    const int m  = lane & 15;
    const int kb = (lane >> 4) * 8;
    const float* p = tile + (size_t)(row0 + m) * 32 + kb;
    float4 a0 = *(const float4*)(p);
    float4 a1 = *(const float4*)(p + 4);
    float4 b0 = *(const float4*)(p + 16);
    float4 b1 = *(const float4*)(p + 20);
    v16h r;
    r[0]=(_Float16)a0.x; r[1]=(_Float16)a0.y; r[2]=(_Float16)a0.z; r[3]=(_Float16)a0.w;
    r[4]=(_Float16)a1.x; r[5]=(_Float16)a1.y; r[6]=(_Float16)a1.z; r[7]=(_Float16)a1.w;
    r[8]=(_Float16)b0.x; r[9]=(_Float16)b0.y; r[10]=(_Float16)b0.z; r[11]=(_Float16)b0.w;
    r[12]=(_Float16)b1.x; r[13]=(_Float16)b1.y; r[14]=(_Float16)b1.z; r[15]=(_Float16)b1.w;
    return r;
}

// A-fragment from an f16 LDS tile [m][LDT]
static __device__ __forceinline__ v16h load_afrag(const _Float16* lds, int row0, int lane) {
    int m  = lane & 15;
    int kb = (lane >> 4) * 8;
    const _Float16* p = lds + (size_t)(row0 + m) * LDT + kb;
    v8h lo = *(const v8h*)(p);
    v8h hi = *(const v8h*)(p + 16);
    return cat8(lo, hi);
}

// B-fragment (32x16) from f16 LDS tile stored transposed [n][k]
static __device__ __forceinline__ v16h load_bfrag(const _Float16* lds, int nt, int lane) {
    int n    = (nt << 4) + (lane & 15);
    int koff = (lane >> 4) * 8;               // lo lanes K0..7/K16..23, hi lanes K8..15/K24..31
    const _Float16* p = lds + (size_t)n * LDT + koff;
    v8h lo = *(const v8h*)(p);
    v8h hi = *(const v8h*)(p + 16);
    return cat8(lo, hi);
}

// =====================================================================
// Generic segment-list GEMM:  out[M,128] = relu(sum_s A_s @ W_seg + b)
// A tiles staged global->LDS with async b128 copies, double-buffered.
// =====================================================================
struct GemmSegs {
    const float* p[4];
    int lda[4];
    int kk[4];
    int nseg;
};

static __device__ __forceinline__ void issue_tileA_async(const float* Ap, int lda, int i0, int k,
                                                         float* buf, int tid) {
    #pragma unroll
    for (int e = 0; e < 4; ++e) {
        int c = e * 256 + tid;          // 1024 chunks of 16B = 128 rows x 32 f32
        int il = c >> 3, jc = c & 7;
        async_load_b128(buf + il * 32 + jc * 4,
                        Ap + (size_t)(i0 + il) * lda + k + jc * 4);
    }
}

__global__ __launch_bounds__(256)
void gemm_relu_kernel(GemmSegs segs, const float* __restrict__ W,
                      const float* __restrict__ bias, float* __restrict__ out,
                      _Float16* __restrict__ out16, int M)
{
    __shared__ __align__(16) float    lds_fa[2][128 * 32];   // async-staged A tiles (f32)
    __shared__ __align__(16) _Float16 lds_w[128 * LDT];      // W tile, transposed [n][k] f16

    const int tid  = threadIdx.x;
    const int wave = tid >> 5;
    const int lane = tid & 31;
    const int i0   = blockIdx.x * 128;

    v8f acc[8];
    #pragma unroll
    for (int nt = 0; nt < 8; ++nt) { v8f z = {0.f,0.f,0.f,0.f,0.f,0.f,0.f,0.f}; acc[nt] = z; }

    // prologue: async-stage first A tile
    issue_tileA_async(segs.p[0], segs.lda[0], i0, 0, lds_fa[0], tid);

    int cur = 0;
    int wrow = 0;
    for (int s = 0; s < segs.nseg; ++s) {
        const int K = segs.kk[s];
        for (int k = 0; k < K; k += 32, wrow += 32) {
            // ---- stage W tile 32x128 : batched load phase, then convert+store ----
            float4 wv[4];
            #pragma unroll
            for (int e = 0; e < 4; ++e) {          // 1024 float4 chunks, coalesced
                int c = e * 256 + tid;
                int kr = c >> 5, nc = c & 31;
                wv[e] = *(const float4*)(W + (size_t)(wrow + kr) * 128 + nc * 4);
            }
            #pragma unroll
            for (int e = 0; e < 4; ++e) {
                int c = e * 256 + tid;
                int kr = c >> 5, nc = c & 31;
                lds_w[(nc * 4 + 0) * LDT + kr] = (_Float16)wv[e].x;
                lds_w[(nc * 4 + 1) * LDT + kr] = (_Float16)wv[e].y;
                lds_w[(nc * 4 + 2) * LDT + kr] = (_Float16)wv[e].z;
                lds_w[(nc * 4 + 3) * LDT + kr] = (_Float16)wv[e].w;
            }
            wait_async0();        // this wave's A-tile chunks have landed
            __syncthreads();      // all waves' chunks + W tile visible

            // issue next A tile into the other buffer (overlaps the WMMAs below)
            int ns = s, nk = k + 32;
            if (nk >= K) { ns = s + 1; nk = 0; }
            if (ns < segs.nseg)
                issue_tileA_async(segs.p[ns], segs.lda[ns], i0, nk, lds_fa[cur ^ 1], tid);

            v16h a = afrag_from_f32(lds_fa[cur], wave * 16, lane);
            #pragma unroll
            for (int nt = 0; nt < 8; ++nt) {
                v16h b = load_bfrag(lds_w, nt, lane);
                acc[nt] = __builtin_amdgcn_wmma_f32_16x16x32_f16(
                    false, a, false, b, (short)0, acc[nt], false, false);
            }
            __syncthreads();
            cur ^= 1;
        }
    }

    // epilogue: bias + relu.  VGPR r -> M = r + 8*(lane>=16), N = nt*16 + (lane&15)
    const int nl = lane & 15;
    const int mh = (lane >> 4) * 8;
    #pragma unroll
    for (int nt = 0; nt < 8; ++nt) {
        const int n = nt * 16 + nl;
        const float bv = bias[n];
        #pragma unroll
        for (int r = 0; r < 8; ++r) {
            const int mm = i0 + wave * 16 + mh + r;
            float v = acc[nt][r] + bv;
            v = v > 0.f ? v : 0.f;
            out[(size_t)mm * 128 + n] = v;
            if (out16) out16[(size_t)mm * 128 + n] = (_Float16)v;
        }
    }
}

// =====================================================================
// Per-node score:  f[i] = X[i,:].dot(w) + b
// =====================================================================
__global__ __launch_bounds__(256)
void score_kernel(const float* __restrict__ X, const float* __restrict__ w,
                  const float* __restrict__ bptr, float* __restrict__ f, int M)
{
    const int wave = threadIdx.x >> 5;
    const int lane = threadIdx.x & 31;
    const int row  = blockIdx.x * 8 + wave;
    if (row >= M) return;
    const float4 xv = ((const float4*)(X + (size_t)row * 128))[lane];
    const float4 wv = ((const float4*)w)[lane];
    float s = xv.x * wv.x + xv.y * wv.y + xv.z * wv.z + xv.w * wv.w;
    #pragma unroll
    for (int off = 16; off; off >>= 1) s += __shfl_xor(s, off, 32);
    if (lane == 0) f[row] = s + bptr[0];
}

static __device__ __forceinline__ float lrelu(float x) { return x > 0.f ? x : 0.02f * x; }

static __device__ __forceinline__ void lse_upd(int msk, float fi, float fj, float& m, float& Z) {
    if (msk) {
        float x = lrelu(fi + fj);
        if (x > m) { Z = Z * __expf(m - x) + 1.f; m = x; }
        else       { Z += __expf(x - m); }
    }
}

// =====================================================================
// Pass A (row-major adj): one wave per row, int4/float4 streaming,
// masked online logsumexp -> m, 1/Z.   NJ must be a multiple of 128.
// =====================================================================
__global__ __launch_bounds__(256)
void softstat_row_kernel(const int* __restrict__ adj, int ldadj,
                         const float* __restrict__ fI, const float* __restrict__ fJ,
                         float* __restrict__ mrow, float* __restrict__ rinv, int NJ)
{
    const int wave = threadIdx.x >> 5;
    const int lane = threadIdx.x & 31;
    const int i = blockIdx.x * 8 + wave;
    const float fi = fI[i];
    const int* arow = adj + (size_t)i * ldadj;
    float m = -INFINITY, Z = 0.f;
    for (int j4 = lane * 4; j4 < NJ; j4 += 128) {        // lane owns 4 consecutive j
        int4   mk = *(const int4*)(arow + j4);
        float4 fj = *(const float4*)(fJ + j4);
        lse_upd(mk.x, fi, fj.x, m, Z);
        lse_upd(mk.y, fi, fj.y, m, Z);
        lse_upd(mk.z, fi, fj.z, m, Z);
        lse_upd(mk.w, fi, fj.w, m, Z);
    }
    #pragma unroll
    for (int off = 16; off; off >>= 1) {
        float mo = __shfl_xor(m, off, 32);
        float Zo = __shfl_xor(Z, off, 32);
        float mn = fmaxf(m, mo);
        float Zn = 0.f;
        if (Z  > 0.f) Zn += Z  * __expf(m  - mn);
        if (Zo > 0.f) Zn += Zo * __expf(mo - mn);
        m = mn; Z = Zn;
    }
    if (lane == 0) {
        mrow[i] = (Z > 0.f) ? m : 0.f;
        rinv[i] = (Z > 0.f) ? 1.f / Z : 0.f;   // empty rows -> zero output (rowmean rule)
    }
}

// =====================================================================
// Pass A (transposed adj): thread per i (coalesced in i), 4-way batched loads
// =====================================================================
__global__ __launch_bounds__(256)
void softstat_col_kernel(const int* __restrict__ adj, int ldadj,
                         const float* __restrict__ fI, const float* __restrict__ fJ,
                         float* __restrict__ mrow, float* __restrict__ rinv, int NJ)
{
    const int i = blockIdx.x * 256 + threadIdx.x;
    const float fi = fI[i];
    float m = -INFINITY, Z = 0.f;
    for (int j = 0; j < NJ; j += 4) {
        // batched independent loads (coalesced across threads)
        int m0 = adj[(size_t)(j + 0) * ldadj + i];
        int m1 = adj[(size_t)(j + 1) * ldadj + i];
        int m2 = adj[(size_t)(j + 2) * ldadj + i];
        int m3 = adj[(size_t)(j + 3) * ldadj + i];
        float4 fj = *(const float4*)(fJ + j);
        lse_upd(m0, fi, fj.x, m, Z);
        lse_upd(m1, fi, fj.y, m, Z);
        lse_upd(m2, fi, fj.z, m, Z);
        lse_upd(m3, fi, fj.w, m, Z);
    }
    mrow[i] = (Z > 0.f) ? m : 0.f;
    rinv[i] = (Z > 0.f) ? 1.f / Z : 0.f;
}

// =====================================================================
// Pass B: out[NI,128] = softmax_row(masked logits) @ FeatJ, via WMMA.
// Adjacency tiles staged with async b128 copies (double-buffered, ASYNCcnt);
// P-tile built from LDS; FeatJ is pre-converted f16 (halves L2 traffic).
// =====================================================================
static __device__ __forceinline__ void issue_adj_async(const int* adj, int ldadj, int transposed,
                                                       int i0, int j0, int* buf, int tid) {
    if (!transposed) {
        // tile layout [il][jl] : 128 x 32 ints
        #pragma unroll
        for (int e = 0; e < 4; ++e) {
            int c = e * 256 + tid;
            int il = c >> 3, jc = c & 7;
            async_load_b128(buf + il * 32 + jc * 4,
                            adj + (size_t)(i0 + il) * ldadj + j0 + jc * 4);
        }
    } else {
        // tile layout [jl][il] : 32 x 128 ints
        #pragma unroll
        for (int e = 0; e < 4; ++e) {
            int c = e * 256 + tid;
            int jl = c >> 5, ic = c & 31;
            async_load_b128(buf + jl * 128 + ic * 4,
                            adj + (size_t)(j0 + jl) * ldadj + i0 + ic * 4);
        }
    }
}

__global__ __launch_bounds__(256)
void attn_pv_kernel(const int* __restrict__ adj, int ldadj, int transposed,
                    const float* __restrict__ fI, const float* __restrict__ fJ,
                    const float* __restrict__ mrow, const float* __restrict__ rinv,
                    const _Float16* __restrict__ FeatJ16, float* __restrict__ out, int NJ)
{
    __shared__ __align__(16) int      lds_adj[2][128 * 32];  // async-staged mask tiles
    __shared__ __align__(16) _Float16 lds_p[128 * LDT];
    __shared__ __align__(16) _Float16 lds_f[128 * LDT];
    __shared__ float s_fi[128];
    __shared__ float s_m[128];
    __shared__ float s_fj[32];

    const int tid  = threadIdx.x;
    const int wave = tid >> 5;
    const int lane = tid & 31;
    const int i0   = blockIdx.x * 128;

    if (tid < 128) { s_fi[tid] = fI[i0 + tid]; s_m[tid] = mrow[i0 + tid]; }

    v8f acc[8];
    #pragma unroll
    for (int nt = 0; nt < 8; ++nt) { v8f z = {0.f,0.f,0.f,0.f,0.f,0.f,0.f,0.f}; acc[nt] = z; }

    issue_adj_async(adj, ldadj, transposed, i0, 0, lds_adj[0], tid);

    int cur = 0;
    for (int j0 = 0; j0 < NJ; j0 += 32) {
        // ---- stage FeatJ tile 32x128 f16 : batched v8h loads, then transpose-store ----
        v8h fv[2];
        #pragma unroll
        for (int e = 0; e < 2; ++e) {          // 512 chunks of 8 f16, coalesced
            int c = e * 256 + tid;
            int kr = c >> 4, nc = c & 15;
            fv[e] = *(const v8h*)(FeatJ16 + (size_t)(j0 + kr) * 128 + nc * 8);
        }
        if (tid < 32) s_fj[tid] = fJ[j0 + tid];
        #pragma unroll
        for (int e = 0; e < 2; ++e) {
            int c = e * 256 + tid;
            int kr = c >> 4, nc = c & 15;
            #pragma unroll
            for (int q = 0; q < 8; ++q)
                lds_f[(nc * 8 + q) * LDT + kr] = fv[e][q];
        }
        wait_async0();
        __syncthreads();

        if (j0 + 32 < NJ)
            issue_adj_async(adj, ldadj, transposed, i0, j0 + 32, lds_adj[cur ^ 1], tid);

        // ---- build P[128][32] f16 : batched mask loads, then exp+store ----
        const int* madj = lds_adj[cur];
        int msk[16];
        if (!transposed) {
            const int jl = tid & 31;
            #pragma unroll
            for (int e = 0; e < 16; ++e) msk[e] = madj[(e * 8 + (tid >> 5)) * 32 + jl];
            const float fj = s_fj[jl];
            #pragma unroll
            for (int e = 0; e < 16; ++e) {
                int il = e * 8 + (tid >> 5);
                float p = 0.f;
                if (msk[e]) p = __expf(lrelu(s_fi[il] + fj) - s_m[il]);
                lds_p[il * LDT + jl] = (_Float16)p;
            }
        } else {
            const int il = tid & 127;
            #pragma unroll
            for (int e = 0; e < 16; ++e) msk[e] = madj[(e * 2 + (tid >> 7)) * 128 + il];
            const float fi = s_fi[il];
            const float mi = s_m[il];
            #pragma unroll
            for (int e = 0; e < 16; ++e) {
                int jl = e * 2 + (tid >> 7);
                float p = 0.f;
                if (msk[e]) p = __expf(lrelu(fi + s_fj[jl]) - mi);
                lds_p[il * LDT + jl] = (_Float16)p;
            }
        }
        __syncthreads();

        v16h a = load_afrag(lds_p, wave * 16, lane);
        #pragma unroll
        for (int nt = 0; nt < 8; ++nt) {
            v16h b = load_bfrag(lds_f, nt, lane);
            acc[nt] = __builtin_amdgcn_wmma_f32_16x16x32_f16(
                false, a, false, b, (short)0, acc[nt], false, false);
        }
        __syncthreads();
        cur ^= 1;
    }

    // epilogue: scale each row by 1/Z
    const int nl = lane & 15;
    const int mh = (lane >> 4) * 8;
    float rv[8];
    #pragma unroll
    for (int r = 0; r < 8; ++r) rv[r] = rinv[i0 + wave * 16 + mh + r];
    #pragma unroll
    for (int nt = 0; nt < 8; ++nt) {
        const int n = nt * 16 + nl;
        #pragma unroll
        for (int r = 0; r < 8; ++r) {
            const int mm = i0 + wave * 16 + mh + r;
            out[(size_t)mm * 128 + n] = acc[nt][r] * rv[r];
        }
    }
}

// =====================================================================
// Host side
// =====================================================================
static const int NA = 8192, NB = 6144, NC = 4096, H = 128;

static void launch_gemm1(const float* A, int lda, int K, const float* W, const float* b,
                         float* out, _Float16* out16, int M, hipStream_t s)
{
    GemmSegs segs{}; segs.p[0] = A; segs.lda[0] = lda; segs.kk[0] = K; segs.nseg = 1;
    gemm_relu_kernel<<<dim3(M / 128), dim3(256), 0, s>>>(segs, W, b, out, out16, M);
}

static void launch_gemm2(const float* A0, const float* A1, const float* W, const float* b,
                         float* out, int M, hipStream_t s)
{
    GemmSegs segs{};
    segs.p[0] = A0; segs.lda[0] = H; segs.kk[0] = H;
    segs.p[1] = A1; segs.lda[1] = H; segs.kk[1] = H;
    segs.nseg = 2;
    gemm_relu_kernel<<<dim3(M / 128), dim3(256), 0, s>>>(segs, W, b, out, nullptr, M);
}

static void launch_gemmN(const float* const* parts, int n, const float* W, const float* b,
                         float* out, int M, hipStream_t s)
{
    GemmSegs segs{};
    for (int i = 0; i < n; ++i) { segs.p[i] = parts[i]; segs.lda[i] = H; segs.kk[i] = H; }
    segs.nseg = n;
    gemm_relu_kernel<<<dim3(M / 128), dim3(256), 0, s>>>(segs, W, b, out, nullptr, M);
}

extern "C" void kernel_launch(void* const* d_in, const int* in_sizes, int n_in,
                              void* d_out, int out_size, void* d_ws, size_t ws_size,
                              hipStream_t stream)
{
    (void)in_sizes; (void)n_in; (void)out_size; (void)ws_size;

    const int*   adj_AB = (const int*)d_in[0];
    const int*   adj_AC = (const int*)d_in[1];
    const int*   adj_BC = (const int*)d_in[2];
    const float* featA  = (const float*)d_in[3];
    const float* featB  = (const float*)d_in[4];
    const float* featC  = (const float*)d_in[5];
    const float* simAin = (const float*)d_in[6];
    const float* simBin = (const float*)d_in[7];
    const float* W_A = (const float*)d_in[8],  *b_A = (const float*)d_in[9];
    const float* W_B = (const float*)d_in[10], *b_B = (const float*)d_in[11];
    const float* W_C = (const float*)d_in[12], *b_C = (const float*)d_in[13];
    const float* W_simA = (const float*)d_in[14], *b_simA = (const float*)d_in[15];
    const float* W_simB = (const float*)d_in[16], *b_simB = (const float*)d_in[17];
    const float* attw = (const float*)d_in[18];
    const float* attb = (const float*)d_in[19];
    const float* Wemb = (const float*)d_in[20];
    const float* bemb = (const float*)d_in[21];
    const float* W_Aemb = (const float*)d_in[22], *b_Aemb = (const float*)d_in[23];
    const float* W_Bemb = (const float*)d_in[24], *b_Bemb = (const float*)d_in[25];
    const float* W_Cemb = (const float*)d_in[26], *b_Cemb = (const float*)d_in[27];

    // bump allocator on workspace (16B aligned chunks)
    char* wp = (char*)d_ws;
    auto allocB = [&](size_t nbytes) -> void* {
        void* r = (void*)wp; wp += (nbytes + 15) & ~(size_t)15; return r;
    };
    auto alloc  = [&](size_t nf) -> float*     { return (float*)allocB(nf * 4); };
    auto alloc16= [&](size_t nf) -> _Float16*  { return (_Float16*)allocB(nf * 2); };

    float* A    = alloc((size_t)NA * H);
    float* B    = alloc((size_t)NB * H);
    float* C    = alloc((size_t)NC * H);
    float* simA = alloc((size_t)NA * H);
    float* simB = alloc((size_t)NB * H);
    _Float16* A16 = alloc16((size_t)NA * H);
    _Float16* B16 = alloc16((size_t)NB * H);
    _Float16* C16 = alloc16((size_t)NC * H);

    const int NIr[6] = {NA, NB, NB, NC, NA, NC};        // AB, BA, BC, CB, AC, CA
    const int NJr[6] = {NB, NA, NC, NB, NC, NA};
    float* sI[6]; float* sJ[6]; float* mr[6]; float* rz[6]; float* av[6];
    for (int l = 0; l < 6; ++l) { sI[l] = alloc(NIr[l]); sJ[l] = alloc(NJr[l]); }
    for (int l = 0; l < 6; ++l) { mr[l] = alloc(NIr[l]); rz[l] = alloc(NIr[l]);
                                  av[l] = alloc((size_t)NIr[l] * H); }
    float* A_from_B = alloc((size_t)NA * H);
    float* B_from_A = alloc((size_t)NB * H);
    float* B_from_C = alloc((size_t)NB * H);
    float* C_from_B = alloc((size_t)NC * H);
    float* A_from_C = alloc((size_t)NA * H);
    float* C_from_A = alloc((size_t)NC * H);

    // ---- 1. input MLPs (emit f16 copies for attention V) + sim layers ----
    launch_gemm1(featA, 512, 512, W_A, b_A, A, A16, NA, stream);
    launch_gemm1(featB, 512, 512, W_B, b_B, B, B16, NB, stream);
    launch_gemm1(featC, 512, 512, W_C, b_C, C, C16, NC, stream);
    launch_gemm1(simAin, NA, NA, W_simA, b_simA, simA, nullptr, NA, stream);
    launch_gemm1(simBin, NB, NB, W_simB, b_simB, simB, nullptr, NB, stream);

    // ---- 2. attention scores ----
    const float* nodeI[6] = {A, B, B, C, A, C};
    const float* nodeJ[6] = {B, A, C, B, C, A};
    const _Float16* nodeJ16[6] = {B16, A16, C16, B16, C16, A16};
    for (int l = 0; l < 6; ++l) {
        score_kernel<<<dim3(NIr[l] / 8), dim3(256), 0, stream>>>(
            nodeI[l], attw + (size_t)(l * 2 + 0) * H, attb + (l * 2 + 0), sI[l], NIr[l]);
        score_kernel<<<dim3(NJr[l] / 8), dim3(256), 0, stream>>>(
            nodeJ[l], attw + (size_t)(l * 2 + 1) * H, attb + (l * 2 + 1), sJ[l], NJr[l]);
    }

    // ---- 3. softmax stats + P@V (flash-style WMMA, async adj staging) ----
    const int*   adjP[6]  = {adj_AB, adj_AB, adj_BC, adj_BC, adj_AC, adj_AC};
    const int    ldadj[6] = {NB, NB, NC, NC, NC, NC};
    const int    trans[6] = {0, 1, 0, 1, 0, 1};
    for (int l = 0; l < 6; ++l) {
        if (!trans[l])
            softstat_row_kernel<<<dim3(NIr[l] / 8), dim3(256), 0, stream>>>(
                adjP[l], ldadj[l], sI[l], sJ[l], mr[l], rz[l], NJr[l]);
        else
            softstat_col_kernel<<<dim3(NIr[l] / 256), dim3(256), 0, stream>>>(
                adjP[l], ldadj[l], sI[l], sJ[l], mr[l], rz[l], NJr[l]);
        attn_pv_kernel<<<dim3(NIr[l] / 128), dim3(256), 0, stream>>>(
            adjP[l], ldadj[l], trans[l], sI[l], sJ[l], mr[l], rz[l],
            nodeJ16[l], av[l], NJr[l]);
    }

    // ---- 4. fusion layers (concat as 2 K-segments) ----
    launch_gemm2(A, av[0], Wemb + 0 * 256 * 128, bemb + 0 * 128, A_from_B, NA, stream);
    launch_gemm2(B, av[1], Wemb + 1 * 256 * 128, bemb + 1 * 128, B_from_A, NB, stream);
    launch_gemm2(B, av[2], Wemb + 2 * 256 * 128, bemb + 2 * 128, B_from_C, NB, stream);
    launch_gemm2(C, av[3], Wemb + 3 * 256 * 128, bemb + 3 * 128, C_from_B, NC, stream);
    launch_gemm2(A, av[4], Wemb + 2 * 256 * 128, bemb + 2 * 128, A_from_C, NA, stream); // bug kept
    launch_gemm2(C, av[5], Wemb + 3 * 256 * 128, bemb + 3 * 128, C_from_A, NC, stream); // bug kept

    // ---- 5. final embeddings straight into d_out ----
    float* out = (float*)d_out;
    float* A_emb = out;
    float* B_emb = out + (size_t)NA * H;
    float* C_emb = out + (size_t)(NA + NB) * H;
    { const float* pa[4] = {simA, A_from_B, A_from_C, A};
      launch_gemmN(pa, 4, W_Aemb, b_Aemb, A_emb, NA, stream); }
    { const float* pb[4] = {simB, B_from_C, B_from_A, B};
      launch_gemmN(pb, 4, W_Bemb, b_Bemb, B_emb, NB, stream); }
    { const float* pc[3] = {C_from_A, C_from_B, C};
      launch_gemmN(pc, 3, W_Cemb, b_Cemb, C_emb, NC, stream); }
}
